// TubeDetrLayer_86157043958255
// MI455X (gfx1250) — matrix operand order repaired and checked by reading, not compiled, not run on hardware
//
#include <hip/hip_runtime.h>

typedef _Float16 half_t;
typedef _Float16 v8h  __attribute__((ext_vector_type(8)));
typedef _Float16 v16h __attribute__((ext_vector_type(16)));
typedef float    v8f  __attribute__((ext_vector_type(8)));

#define LDSP 40   // padded LDS row stride in halves (80B -> bank-conflict free, 16B aligned)

// ---------------------------------------------------------------------------
// Generic batched WMMA GEMM:  C = alpha * A[M,K] x B[K,N] (+bias)(+resid)(gelu)
//  A_F32 : A is fp32 (converted to f16 while staging to LDS), else f16
//  B_NT  : B operand given transposed as Bt[N,K] row-major (weights / K^T),
//          else B[K,N] row-major (transposed into LDS at staging)
//  Block tile: 128(M) x 64(N), 8 waves, each wave = one 16-row strip x 4 tiles
//  Batch z decomposed as (b = z/H, h = z%H) with per-b / per-h strides.
// ---------------------------------------------------------------------------
template<bool A_F32, bool B_NT, bool GELU>
__global__ __launch_bounds__(256)
void gemm_wmma(const void* __restrict__ Aptr, const half_t* __restrict__ Bptr,
               const float* __restrict__ bias, const float* __restrict__ resid,
               float* __restrict__ Cf, half_t* __restrict__ Ch,
               int M, int N, int K, int lda, int ldb, int ldc, float alpha, int H,
               long long aSb, long long aSh, long long bSb, long long bSh,
               long long cSb, long long cSh)
{
    __shared__ half_t lA[128 * LDSP];
    __shared__ half_t lB[64 * LDSP];

    const int z  = blockIdx.z;
    const int bi = z / H, hi = z % H;
    const long long aBase = (long long)bi * aSb + (long long)hi * aSh;
    const long long bBase = (long long)bi * bSb + (long long)hi * bSh;
    const long long cBase = (long long)bi * cSb + (long long)hi * cSh;

    const int m0 = blockIdx.x * 128;
    const int n0 = blockIdx.y * 64;
    const int tid = threadIdx.x;
    const int w = tid >> 5, l = tid & 31;

    v8f acc[4] = {};

    for (int k0 = 0; k0 < K; k0 += 32) {
        // ---- stage A tile: 128 rows x 32 halves ----
        {
            const int row = tid >> 1, hf = tid & 1;
            const long long off = aBase + (long long)(m0 + row) * lda + k0 + hf * 16;
            half_t* dst = &lA[row * LDSP + hf * 16];
            if constexpr (A_F32) {
                const float* Af = (const float*)Aptr + off;
#pragma unroll
                for (int i = 0; i < 16; ++i) dst[i] = (half_t)Af[i];
            } else {
                const half_t* Ah = (const half_t*)Aptr + off;
                *(v8h*)dst       = *(const v8h*)Ah;
                *(v8h*)(dst + 8) = *(const v8h*)(Ah + 8);
            }
        }
        // ---- stage B tile into lB[n][k] : 64 x 32 halves ----
        if (tid < 128) {
            if constexpr (B_NT) {
                const int row = tid >> 1, hf = tid & 1;   // row = n
                const half_t* src = Bptr + bBase + (long long)(n0 + row) * ldb + k0 + hf * 16;
                half_t* dst = &lB[row * LDSP + hf * 16];
                *(v8h*)dst       = *(const v8h*)src;
                *(v8h*)(dst + 8) = *(const v8h*)(src + 8);
            } else {
                const int kr = tid >> 2, q = tid & 3;     // row = k, 16 n values
                const half_t* src = Bptr + bBase + (long long)(k0 + kr) * ldb + n0 + q * 16;
#pragma unroll
                for (int i = 0; i < 16; ++i) lB[(q * 16 + i) * LDSP + kr] = src[i];
            }
        }
        __syncthreads();

        // ---- fragments + WMMA ----
        const int mrow = w * 16 + (l & 15);
        const int hh   = l >> 4;
        v8h a_lo = *(const v8h*)&lA[mrow * LDSP + hh * 8];
        v8h a_hi = *(const v8h*)&lA[mrow * LDSP + 16 + hh * 8];
        v16h a = __builtin_shufflevector(a_lo, a_hi, 0,1,2,3,4,5,6,7,8,9,10,11,12,13,14,15);
#pragma unroll
        for (int c = 0; c < 4; ++c) {
            const int nl = c * 16 + (l & 15);
            const int ks = hh * 8;
            v8h b_lo = *(const v8h*)&lB[nl * LDSP + ks];
            v8h b_hi = *(const v8h*)&lB[nl * LDSP + 16 + ks];
            v16h b = __builtin_shufflevector(b_lo, b_hi, 0,1,2,3,4,5,6,7,8,9,10,11,12,13,14,15);
            acc[c] = __builtin_amdgcn_wmma_f32_16x16x32_f16(
                false, a, false, b, (short)0, acc[c], false, false);
        }
        __syncthreads();
    }

    // ---- epilogue ----
#pragma unroll
    for (int c = 0; c < 4; ++c) {
#pragma unroll
        for (int i = 0; i < 8; ++i) {
            const int m = m0 + w * 16 + (l >> 4) * 8 + i;
            const int n = n0 + c * 16 + (l & 15);
            float v = acc[c][i] * alpha;
            if (bias) v += bias[n];
            const long long coff = cBase + (long long)m * ldc + n;
            if (resid) v += resid[coff];
            if constexpr (GELU) v = 0.5f * v * (1.0f + erff(v * 0.70710678118654752f));
            if (Cf) Cf[coff] = v;
            if (Ch) Ch[coff] = (half_t)v;
        }
    }
}

// ---- transpose+convert weight: W[K,N] fp32 -> Wt[N,K] f16 ----
__global__ void wtrans(const float* __restrict__ W, half_t* __restrict__ Wt, int K, int N)
{
    long long i = (long long)blockIdx.x * blockDim.x + threadIdx.x;
    if (i >= (long long)K * N) return;
    int k = (int)(i / N), n = (int)(i % N);
    Wt[(long long)n * K + k] = (half_t)W[i];
}

// ---- softmax over rows of 256 (self-attn scores), fp32 in -> f16 out ----
__global__ __launch_bounds__(256)
void softmax256(const float* __restrict__ in, half_t* __restrict__ out)
{
    const long long row = blockIdx.x;
    const int t = threadIdx.x;
    __shared__ float s[256];
    float v = in[row * 256 + t];
    s[t] = v; __syncthreads();
    for (int st = 128; st > 0; st >>= 1) { if (t < st) s[t] = fmaxf(s[t], s[t + st]); __syncthreads(); }
    float mx = s[0]; __syncthreads();
    float e = __expf(v - mx);
    s[t] = e; __syncthreads();
    for (int st = 128; st > 0; st >>= 1) { if (t < st) s[t] += s[t + st]; __syncthreads(); }
    out[row * 256 + t] = (half_t)(e / s[0]);
}

// ---- LayerNorm over rows of 256 (writes fp32 and optional f16 copy) ----
__global__ __launch_bounds__(256)
void layernorm256(const float* __restrict__ in, const float* __restrict__ g,
                  const float* __restrict__ b, float* __restrict__ outf,
                  half_t* __restrict__ outh)
{
    const long long row = blockIdx.x;
    const int t = threadIdx.x;
    __shared__ float s1[256], s2[256];
    float v = in[row * 256 + t];
    s1[t] = v; s2[t] = v * v; __syncthreads();
    for (int st = 128; st > 0; st >>= 1) {
        if (t < st) { s1[t] += s1[t + st]; s2[t] += s2[t + st]; }
        __syncthreads();
    }
    float mu  = s1[0] * (1.0f / 256.0f);
    float var = s2[0] * (1.0f / 256.0f) - mu * mu;
    float o = (v - mu) * rsqrtf(var + 1e-5f) * g[t] + b[t];
    if (outf) outf[row * 256 + t] = o;
    if (outh) outh[row * 256 + t] = (half_t)o;
}

// ---- block-diagonal cross-attention: query (b,t) attends to its 64 feat rows.
// One 256-thread block per (b,t); threads = (h in 0..3) x (j in 0..63).
__global__ __launch_bounds__(256)
void cross_attn(const float* __restrict__ qc, const half_t* __restrict__ kf,
                const half_t* __restrict__ vf, half_t* __restrict__ out)
{
    const long long bt = blockIdx.x;
    const int tid = threadIdx.x, h = tid >> 6, j = tid & 63;
    __shared__ float sq[256], sc[256], red[256];
    sq[tid] = qc[bt * 256 + tid];
    __syncthreads();

    // scores: j = key index
    const half_t* krow = kf + (bt * 64 + j) * 256 + h * 64;
    float acc = 0.0f;
#pragma unroll 8
    for (int d = 0; d < 64; ++d) acc += sq[h * 64 + d] * (float)krow[d];
    acc *= 0.125f;                       // 1/sqrt(DH)

    red[tid] = acc; __syncthreads();
    for (int st = 32; st > 0; st >>= 1) { if (j < st) red[tid] = fmaxf(red[tid], red[tid + st]); __syncthreads(); }
    float mx = red[h << 6]; __syncthreads();
    float e = __expf(acc - mx);
    red[tid] = e; __syncthreads();
    for (int st = 32; st > 0; st >>= 1) { if (j < st) red[tid] += red[tid + st]; __syncthreads(); }
    float sm = red[h << 6]; __syncthreads();
    sc[tid] = e / sm;
    __syncthreads();

    // output: j = head-dim index
    float o = 0.0f;
#pragma unroll 8
    for (int n = 0; n < 64; ++n)
        o += sc[h * 64 + n] * (float)vf[(bt * 64 + n) * 256 + h * 64 + j];
    out[bt * 256 + h * 64 + j] = (half_t)o;
}

// ---------------------------------------------------------------------------
extern "C" void kernel_launch(void* const* d_in, const int* in_sizes, int n_in,
                              void* d_out, int out_size, void* d_ws, size_t ws_size,
                              hipStream_t stream)
{
    const int Cc = 256, T = 256, Bb = 16, Hh = 4, FFd = 1024;
    const int M1 = Bb * T;            // 4096
    const int M2 = Bb * T * 64;       // 262144 feat rows

    const float* emb    = (const float*)d_in[0];
    const float* feat   = (const float*)d_in[1];
    const float* Wq_a   = (const float*)d_in[2];  const float* bq_a  = (const float*)d_in[3];
    const float* Wk_a   = (const float*)d_in[4];  const float* bk_a  = (const float*)d_in[5];
    const float* Wv_a   = (const float*)d_in[6];  const float* bv_a  = (const float*)d_in[7];
    const float* Wo_a   = (const float*)d_in[8];  const float* bo_a  = (const float*)d_in[9];
    const float* Wqkv_s = (const float*)d_in[10]; const float* bqkv_s= (const float*)d_in[11];
    const float* Wo_s   = (const float*)d_in[12]; const float* bo_s  = (const float*)d_in[13];
    const float* Wqkv_c = (const float*)d_in[14]; const float* bqkv_c= (const float*)d_in[15];
    const float* Wo_c   = (const float*)d_in[16]; const float* bo_c  = (const float*)d_in[17];
    const float* W1     = (const float*)d_in[18]; const float* b1    = (const float*)d_in[19];
    const float* W2     = (const float*)d_in[20]; const float* b2    = (const float*)d_in[21];
    const float* g1 = (const float*)d_in[22]; const float* be1 = (const float*)d_in[23];
    const float* g2 = (const float*)d_in[24]; const float* be2 = (const float*)d_in[25];
    const float* g3 = (const float*)d_in[26]; const float* be3 = (const float*)d_in[27];
    float* outp = (float*)d_out;

    // ---- workspace bump allocator ----
    char* p = (char*)d_ws;
    auto alloc = [&](size_t bytes) -> char* {
        char* r = p; p += (bytes + 255) & ~(size_t)255; return r;
    };
    const size_t CC = (size_t)Cc * Cc;               // 65536
    half_t* WqTa = (half_t*)alloc(CC * 2);
    half_t* WkTa = (half_t*)alloc(CC * 2);
    half_t* WvTa = (half_t*)alloc(CC * 2);
    half_t* WoTa = (half_t*)alloc(CC * 2);
    half_t* WsT0 = (half_t*)alloc(CC * 2);
    half_t* WsT1 = (half_t*)alloc(CC * 2);
    half_t* WsT2 = (half_t*)alloc(CC * 2);
    half_t* WoTs = (half_t*)alloc(CC * 2);
    half_t* WcT0 = (half_t*)alloc(CC * 2);
    half_t* WcT1 = (half_t*)alloc(CC * 2);
    half_t* WcT2 = (half_t*)alloc(CC * 2);
    half_t* WoTc = (half_t*)alloc(CC * 2);
    half_t* W1T  = (half_t*)alloc((size_t)Cc * FFd * 2);
    half_t* W2T  = (half_t*)alloc((size_t)FFd * Cc * 2);

    const size_t MC = (size_t)M1 * Cc;               // 4096*256
    half_t* qh   = (half_t*)alloc(MC * 2);
    half_t* kh   = (half_t*)alloc(MC * 2);
    half_t* vh   = (half_t*)alloc(MC * 2);
    half_t* attnh= (half_t*)alloc((size_t)Bb * Hh * T * T * 2);
    half_t* aoh  = (half_t*)alloc(MC * 2);
    half_t* x1h  = (half_t*)alloc(MC * 2);
    half_t* x2h  = (half_t*)alloc(MC * 2);
    half_t* x3h  = (half_t*)alloc(MC * 2);
    half_t* coh  = (half_t*)alloc(MC * 2);
    half_t* h1h  = (half_t*)alloc((size_t)M1 * FFd * 2);
    half_t* kfh  = (half_t*)alloc((size_t)M2 * Cc * 2);
    half_t* vfh  = (half_t*)alloc((size_t)M2 * Cc * 2);

    float* scoresf = (float*)alloc((size_t)Bb * Hh * T * T * 4);
    float* x1  = (float*)alloc(MC * 4);
    float* yb  = (float*)alloc(MC * 4);
    float* x2  = (float*)alloc(MC * 4);
    float* qcf = (float*)alloc(MC * 4);
    float* z2  = (float*)alloc(MC * 4);
    float* x3  = (float*)alloc(MC * 4);
    float* z3  = (float*)alloc(MC * 4);

    // ---- weight transpose/convert ----
    wtrans<<<256, 256, 0, stream>>>(Wq_a, WqTa, Cc, Cc);
    wtrans<<<256, 256, 0, stream>>>(Wk_a, WkTa, Cc, Cc);
    wtrans<<<256, 256, 0, stream>>>(Wv_a, WvTa, Cc, Cc);
    wtrans<<<256, 256, 0, stream>>>(Wo_a, WoTa, Cc, Cc);
    wtrans<<<256, 256, 0, stream>>>(Wqkv_s + 0 * CC, WsT0, Cc, Cc);
    wtrans<<<256, 256, 0, stream>>>(Wqkv_s + 1 * CC, WsT1, Cc, Cc);
    wtrans<<<256, 256, 0, stream>>>(Wqkv_s + 2 * CC, WsT2, Cc, Cc);
    wtrans<<<256, 256, 0, stream>>>(Wo_s, WoTs, Cc, Cc);
    wtrans<<<256, 256, 0, stream>>>(Wqkv_c + 0 * CC, WcT0, Cc, Cc);
    wtrans<<<256, 256, 0, stream>>>(Wqkv_c + 1 * CC, WcT1, Cc, Cc);
    wtrans<<<256, 256, 0, stream>>>(Wqkv_c + 2 * CC, WcT2, Cc, Cc);
    wtrans<<<256, 256, 0, stream>>>(Wo_c, WoTc, Cc, Cc);
    wtrans<<<1024, 256, 0, stream>>>(W1, W1T, Cc, FFd);
    wtrans<<<1024, 256, 0, stream>>>(W2, W2T, FFd, Cc);

    const dim3 gP(32, 4, 1);        // 4096 x 256
    const dim3 gS(2, 4, 64);        // per-(b,h) 256x256, K=64
    const dim3 gAV(2, 1, 64);       // per-(b,h) 256x64,  K=256
    const dim3 gF(2048, 4, 1);      // 262144 x 256
    const dim3 gFF1(32, 16, 1);     // 4096 x 1024
    const long long SQ = (long long)T * Cc;      // 65536
    const long long ST = (long long)T * T;       // 65536

    // ===== stage 1: x1 = emb + MHA(emb) =====
    gemm_wmma<true , true , false><<<gP, 256, 0, stream>>>(emb, WqTa, bq_a, nullptr, nullptr, qh,
        M1, Cc, Cc, Cc, Cc, Cc, 1.0f, 1, 0,0,0,0,0,0);
    gemm_wmma<true , true , false><<<gP, 256, 0, stream>>>(emb, WkTa, bk_a, nullptr, nullptr, kh,
        M1, Cc, Cc, Cc, Cc, Cc, 1.0f, 1, 0,0,0,0,0,0);
    gemm_wmma<true , true , false><<<gP, 256, 0, stream>>>(emb, WvTa, bv_a, nullptr, nullptr, vh,
        M1, Cc, Cc, Cc, Cc, Cc, 1.0f, 1, 0,0,0,0,0,0);
    gemm_wmma<false, true , false><<<gS, 256, 0, stream>>>(qh, kh, nullptr, nullptr, scoresf, nullptr,
        T, T, 64, Cc, Cc, T, 0.125f, Hh, SQ, 64, SQ, 64, (long long)Hh*ST, ST);
    softmax256<<<Bb * Hh * T, 256, 0, stream>>>(scoresf, attnh);
    gemm_wmma<false, false, false><<<gAV, 256, 0, stream>>>(attnh, vh, nullptr, nullptr, nullptr, aoh,
        T, 64, T, T, Cc, Cc, 1.0f, Hh, (long long)Hh*ST, ST, SQ, 64, SQ, 64);
    gemm_wmma<false, true , false><<<gP, 256, 0, stream>>>(aoh, WoTa, bo_a, emb, x1, x1h,
        M1, Cc, Cc, Cc, Cc, Cc, 1.0f, 1, 0,0,0,0,0,0);

    // ===== stage 2a: x2 = LN1(x1 + SelfAttn(x1)) =====
    gemm_wmma<false, true , false><<<gP, 256, 0, stream>>>(x1h, WsT0, bqkv_s + 0 * Cc, nullptr, nullptr, qh,
        M1, Cc, Cc, Cc, Cc, Cc, 1.0f, 1, 0,0,0,0,0,0);
    gemm_wmma<false, true , false><<<gP, 256, 0, stream>>>(x1h, WsT1, bqkv_s + 1 * Cc, nullptr, nullptr, kh,
        M1, Cc, Cc, Cc, Cc, Cc, 1.0f, 1, 0,0,0,0,0,0);
    gemm_wmma<false, true , false><<<gP, 256, 0, stream>>>(x1h, WsT2, bqkv_s + 2 * Cc, nullptr, nullptr, vh,
        M1, Cc, Cc, Cc, Cc, Cc, 1.0f, 1, 0,0,0,0,0,0);
    gemm_wmma<false, true , false><<<gS, 256, 0, stream>>>(qh, kh, nullptr, nullptr, scoresf, nullptr,
        T, T, 64, Cc, Cc, T, 0.125f, Hh, SQ, 64, SQ, 64, (long long)Hh*ST, ST);
    softmax256<<<Bb * Hh * T, 256, 0, stream>>>(scoresf, attnh);
    gemm_wmma<false, false, false><<<gAV, 256, 0, stream>>>(attnh, vh, nullptr, nullptr, nullptr, aoh,
        T, 64, T, T, Cc, Cc, 1.0f, Hh, (long long)Hh*ST, ST, SQ, 64, SQ, 64);
    gemm_wmma<false, true , false><<<gP, 256, 0, stream>>>(aoh, WoTs, bo_s, x1, yb, nullptr,
        M1, Cc, Cc, Cc, Cc, Cc, 1.0f, 1, 0,0,0,0,0,0);
    layernorm256<<<M1, 256, 0, stream>>>(yb, g1, be1, x2, x2h);

    // ===== stage 2b: x3 = LN2(x2 + CrossAttn(x2, feat)) =====
    gemm_wmma<false, true , false><<<gP, 256, 0, stream>>>(x2h, WcT0, bqkv_c + 0 * Cc, nullptr, qcf, nullptr,
        M1, Cc, Cc, Cc, Cc, Cc, 1.0f, 1, 0,0,0,0,0,0);
    gemm_wmma<true , true , false><<<gF, 256, 0, stream>>>(feat, WcT1, bqkv_c + 1 * Cc, nullptr, nullptr, kfh,
        M2, Cc, Cc, Cc, Cc, Cc, 1.0f, 1, 0,0,0,0,0,0);
    gemm_wmma<true , true , false><<<gF, 256, 0, stream>>>(feat, WcT2, bqkv_c + 2 * Cc, nullptr, nullptr, vfh,
        M2, Cc, Cc, Cc, Cc, Cc, 1.0f, 1, 0,0,0,0,0,0);
    cross_attn<<<M1, 256, 0, stream>>>(qcf, kfh, vfh, coh);
    gemm_wmma<false, true , false><<<gP, 256, 0, stream>>>(coh, WoTc, bo_c, x2, z2, nullptr,
        M1, Cc, Cc, Cc, Cc, Cc, 1.0f, 1, 0,0,0,0,0,0);
    layernorm256<<<M1, 256, 0, stream>>>(z2, g2, be2, x3, x3h);

    // ===== stage 2c: out = LN3(x3 + W2·gelu(W1·x3)) =====
    gemm_wmma<false, true , true ><<<gFF1, 256, 0, stream>>>(x3h, W1T, b1, nullptr, nullptr, h1h,
        M1, FFd, Cc, Cc, Cc, FFd, 1.0f, 1, 0,0,0,0,0,0);
    gemm_wmma<false, true , false><<<gP, 256, 0, stream>>>(h1h, W2T, b2, x3, z3, nullptr,
        M1, Cc, FFd, FFd, FFd, Cc, 1.0f, 1, 0,0,0,0,0,0);
    layernorm256<<<M1, 256, 0, stream>>>(z3, g3, be3, outp, nullptr);

    (void)in_sizes; (void)n_in; (void)out_size; (void)ws_size;
}
